// MaskedSelfAttention_11828339933829
// MI455X (gfx1250) — compile-verified, matrix operand order checked by reading, not run on hardware
//
#include <hip/hip_runtime.h>
#include <hip/hip_bf16.h>
#include <cstdint>
#include <cstddef>

typedef __attribute__((ext_vector_type(16))) _Float16 v16h;
typedef __attribute__((ext_vector_type(8)))  _Float16 v8h;
typedef __attribute__((ext_vector_type(4)))  _Float16 v4h;
typedef __attribute__((ext_vector_type(8)))  float    v8f;

#define B_    4
#define S_    2048
#define E_    1024
#define KD_   1024
#define NROW  (B_ * S_)          // 8192 total rows
#define SCALE_ 0.03125f          // 1/sqrt(1024)

// ---------------------------------------------------------------------------
// WMMA helpers (layouts per CDNA5 ISA 7.12.2)
// A (16x32 f16): lane<16 -> row=lane, elems j: K = j (j<8), 16+j-8 (j>=8)
//                lane>=16 -> row=lane-16, K = 8+j / 24+(j-8)
// B (32x16 f16): same pattern with N = lane%16 in place of M
// C/D (16x16 f32): elem r -> row r + (lane<16?0:8), col = lane%16
// ---------------------------------------------------------------------------
__device__ __forceinline__ v8f wmma16(v16h a, v16h b, v8f c) {
  return __builtin_amdgcn_wmma_f32_16x16x32_f16(false, a, false, b,
                                                (short)0, c, false, false);
}

__device__ __forceinline__ v8f vzero8() {
  v8f z;
#pragma unroll
  for (int i = 0; i < 8; ++i) z[i] = 0.0f;
  return z;
}

__device__ __forceinline__ v16h pack16(v8h lo, v8h hi) {
  v16h r;
#pragma unroll
  for (int i = 0; i < 8; ++i) { r[i] = lo[i]; r[i + 8] = hi[i]; }
  return r;
}

// Load an f16 fragment: 8 halves at p, 8 halves at p+16 (16B aligned each)
__device__ __forceinline__ v16h load_frag_f16(const _Float16* __restrict__ p) {
  v8h lo = *(const v8h*)(p);
  v8h hi = *(const v8h*)(p + 16);
  return pack16(lo, hi);
}

// ---------------------------------------------------------------------------
// Kernel 0: fp32 -> f16 conversion pre-pass (pure bandwidth, runs once).
// ---------------------------------------------------------------------------
__global__ __launch_bounds__(256) void cvt_f16_kernel(
    const float* __restrict__ src, _Float16* __restrict__ dst, int n4) {
  const int i = blockIdx.x * 256 + threadIdx.x;
  if (i < n4) {
    const float4 v = ((const float4*)src)[i];
    v4h h;
    h[0] = (_Float16)v.x; h[1] = (_Float16)v.y;
    h[2] = (_Float16)v.z; h[3] = (_Float16)v.w;
    ((v4h*)dst)[i] = h;
  }
}

// ---------------------------------------------------------------------------
// Kernel 1: QKV projection.  q/k/v[m,n] = sum_e x[m,e] * W[n,e]  (f16 in,
// f32 WMMA accumulate).  One wave -> 16-row M tile x 128 columns (8 D tiles).
// Explicit register ping-pong: loads for E-step s+1 issue before the 8-WMMA
// burst of step s, so VMEM latency overlaps matrix math inside one wave.
// Tasks: which(3) * mt(512) * ntg(8) = 12288, 8 waves/block -> 1536 blocks.
// ---------------------------------------------------------------------------
__global__ __launch_bounds__(256) void qkv_proj_kernel(
    const _Float16* __restrict__ xh,
    const _Float16* __restrict__ WQh, const _Float16* __restrict__ WKh,
    const _Float16* __restrict__ WVh,
    _Float16* __restrict__ Qh, _Float16* __restrict__ Kh,
    _Float16* __restrict__ Vt) {
  const int lane  = threadIdx.x & 31;
  const int wave  = threadIdx.x >> 5;
  const int task  = blockIdx.x * 8 + wave;          // 0..12287
  const int which = task / (512 * 8);               // 0=Q 1=K 2=V
  const int rem   = task % (512 * 8);
  const int mt    = rem / 8;                        // 0..511
  const int ntg   = rem % 8;                        // column group *128

  const _Float16* __restrict__ W =
      (which == 0) ? WQh : ((which == 1) ? WKh : WVh);

  const int lhalf = lane & 15;
  const int kbase = (lane < 16) ? 0 : 8;
  const int arow  = mt * 16 + lhalf;                // global row in [0,8192)

  const _Float16* __restrict__ ap = xh + (size_t)arow * E_ + kbase;
  const _Float16* __restrict__ wp =
      W + (size_t)(ntg * 128 + lhalf) * E_ + kbase;

  v8f acc[8];
#pragma unroll
  for (int t = 0; t < 8; ++t) acc[t] = vzero8();

  // ---- prologue: load phase 0 ----
  v16h afA = load_frag_f16(ap);
  v16h bfA[8];
#pragma unroll
  for (int t = 0; t < 8; ++t) bfA[t] = load_frag_f16(wp + (size_t)t * 16 * E_);

  // 32 E-steps total, processed 2 per trip (ping-pong A/B buffers)
  for (int e0 = 0; e0 < E_; e0 += 64) {
    // issue loads for step e0+32 (always valid: last trip e0=960 -> 992)
    v16h afB = load_frag_f16(ap + e0 + 32);
    v16h bfB[8];
#pragma unroll
    for (int t = 0; t < 8; ++t)
      bfB[t] = load_frag_f16(wp + (size_t)t * 16 * E_ + e0 + 32);

    // WMMA burst for step e0 (uses A buffers; B-buffer loads in flight)
#pragma unroll
    for (int t = 0; t < 8; ++t) acc[t] = wmma16(afA, bfA[t], acc[t]);

    // issue loads for step e0+64 (guarded on final trip)
    if (e0 + 64 < E_) {
      afA = load_frag_f16(ap + e0 + 64);
#pragma unroll
      for (int t = 0; t < 8; ++t)
        bfA[t] = load_frag_f16(wp + (size_t)t * 16 * E_ + e0 + 64);
    }

    // WMMA burst for step e0+32
#pragma unroll
    for (int t = 0; t < 8; ++t) acc[t] = wmma16(afB, bfB[t], acc[t]);
  }

  const int rbase = (lane < 16) ? 0 : 8;
#pragma unroll
  for (int t = 0; t < 8; ++t) {
    const int col = ntg * 128 + t * 16 + lhalf;
#pragma unroll
    for (int r = 0; r < 8; ++r) {
      const int row = mt * 16 + r + rbase;          // global row
      const _Float16 hv = (_Float16)acc[t][r];
      if (which == 0) {
        Qh[(size_t)row * KD_ + col] = hv;
      } else if (which == 1) {
        Kh[(size_t)row * KD_ + col] = hv;
      } else {
        const int b = row >> 11;                    // /2048
        const int s = row & (S_ - 1);
        Vt[((size_t)b * KD_ + col) * S_ + s] = hv;  // per-batch transpose
      }
    }
  }
}

// ---------------------------------------------------------------------------
// Kernel 2: flash attention (causal).  One block = (batch, 16-query block).
// 8 waves; wave w owns feature/value slice [128w, 128w+128).
// V fragments are loaded at the TOP of each key iteration: they are not
// consumed until after three barriers + softmax, so their latency is fully
// hidden behind the score WMMAs and LDS phases.
// ---------------------------------------------------------------------------
__global__ __launch_bounds__(256) void flash_attn_kernel(
    const _Float16* __restrict__ Qh, const _Float16* __restrict__ Kh,
    const _Float16* __restrict__ Vt, float* __restrict__ out) {
  __shared__ __align__(16) float    lds_part[8][16][32];  // 16 KB
  __shared__ __align__(16) float    lds_s[16][32];        //  2 KB
  __shared__ __align__(16) _Float16 lds_p[16][32];        //  1 KB
  __shared__ float lds_alpha[16];
  __shared__ float lds_l[16];

  const int batch = blockIdx.x >> 7;          // /128
  const int qb    = blockIdx.x & 127;         // query block
  const int lane  = threadIdx.x & 31;
  const int wave  = threadIdx.x >> 5;
  const int lhalf = lane & 15;
  const int kbase = (lane < 16) ? 0 : 8;
  const int rbase = (lane < 16) ? 0 : 8;
  const int kb    = wave * 128;               // this wave's feature slice

  // Preload Q fragments (16 rows x 128 features -> 4 A-fragments)
  const int grow = batch * S_ + qb * 16 + lhalf;
  v16h qf[4];
#pragma unroll
  for (int i = 0; i < 4; ++i)
    qf[i] = load_frag_f16(Qh + (size_t)grow * KD_ + kb + i * 32 + kbase);

  // Per-lane base pointers for K (keys lhalf / lhalf+16) and V columns.
  const _Float16* __restrict__ kp0 =
      Kh + (size_t)(batch * S_ + lhalf) * KD_ + kb + kbase;
  const _Float16* __restrict__ kp1 = kp0 + (size_t)16 * KD_;
  const _Float16* __restrict__ vp =
      Vt + ((size_t)batch * KD_ + kb + lhalf) * S_ + kbase;

  v8f oacc[8];
#pragma unroll
  for (int t = 0; t < 8; ++t) oacc[t] = vzero8();

  float m_run = -__builtin_inff();   // live in lanes handling softmax rows
  float l_run = 0.0f;

  const int nIter = (qb * 16 + 15) / 32 + 1;  // 32-key steps (causal bound)
  for (int it = 0; it < nIter; ++it) {
    const int k0 = it * 32;                   // key base within batch

    // Prefetch next iteration's K/V lines (global_prefetch_b8).
    if (it + 1 < nIter) {
      __builtin_prefetch(kp0 + (size_t)(k0 + 32) * KD_, 0, 3);
      __builtin_prefetch(kp1 + (size_t)(k0 + 32) * KD_, 0, 3);
      __builtin_prefetch(vp + k0 + 32, 0, 3);
    }

    // ---- issue ALL loads for this iteration up front ----
    // K fragments (needed by the score WMMAs just below)
    v16h bk0[4], bk1[4];
#pragma unroll
    for (int i = 0; i < 4; ++i) {
      bk0[i] = load_frag_f16(kp0 + (size_t)k0 * KD_ + i * 32);
      bk1[i] = load_frag_f16(kp1 + (size_t)k0 * KD_ + i * 32);
    }
    // V fragments (not consumed until after softmax -> latency fully hidden)
    v16h vf[8];
#pragma unroll
    for (int t = 0; t < 8; ++t)
      vf[t] = load_frag_f16(vp + (size_t)t * 16 * S_ + k0);

    // ---- partial scores: 8 back-to-back WMMAs ----
    v8f c0 = vzero8(), c1 = vzero8();
#pragma unroll
    for (int i = 0; i < 4; ++i) c0 = wmma16(qf[i], bk0[i], c0);
#pragma unroll
    for (int i = 0; i < 4; ++i) c1 = wmma16(qf[i], bk1[i], c1);

#pragma unroll
    for (int r = 0; r < 8; ++r) {
      const int row = r + rbase;
      lds_part[wave][row][lhalf]      = c0[r];
      lds_part[wave][row][16 + lhalf] = c1[r];
    }
    __syncthreads();

    // ---- cross-wave reduction + scale + causal mask ----
    for (int idx = threadIdx.x; idx < 512; idx += 256) {
      const int row = idx >> 5, col = idx & 31;
      float s = 0.0f;
#pragma unroll
      for (int w = 0; w < 8; ++w) s += lds_part[w][row][col];
      s *= SCALE_;
      const int kg = k0 + col;
      const int qg = qb * 16 + row;
      if (kg > qg) s = -__builtin_inff();
      lds_s[row][col] = s;
    }
    __syncthreads();

    // ---- online softmax (one thread per query row) ----
    if (threadIdx.x < 16) {
      const int r = threadIdx.x;
      float mx = m_run;
#pragma unroll
      for (int c = 0; c < 32; ++c) mx = fmaxf(mx, lds_s[r][c]);
      const float alpha = __expf(m_run - mx);   // 0 on first iteration
      float rs = 0.0f;
#pragma unroll
      for (int c = 0; c < 32; ++c) {
        const float p = __expf(lds_s[r][c] - mx);
        rs += p;
        lds_p[r][c] = (_Float16)p;
      }
      l_run = l_run * alpha + rs;
      m_run = mx;
      lds_alpha[r] = alpha;
      lds_l[r]     = l_run;
    }
    __syncthreads();

    // ---- rescale accumulators, then O += P x V (K = 32 keys) ----
    float alv[8];
#pragma unroll
    for (int r = 0; r < 8; ++r) alv[r] = lds_alpha[r + rbase];
#pragma unroll
    for (int t = 0; t < 8; ++t)
#pragma unroll
      for (int r = 0; r < 8; ++r) oacc[t][r] *= alv[r];

    const v16h pf = pack16(*(const v8h*)&lds_p[lhalf][kbase],
                           *(const v8h*)&lds_p[lhalf][16 + kbase]);
#pragma unroll
    for (int t = 0; t < 8; ++t) oacc[t] = wmma16(pf, vf[t], oacc[t]);
  }

  // ---- normalize by l and write fp32 output ----
  float lv[8];
#pragma unroll
  for (int r = 0; r < 8; ++r) lv[r] = lds_l[r + rbase];
#pragma unroll
  for (int t = 0; t < 8; ++t) {
    const int col = kb + t * 16 + lhalf;
#pragma unroll
    for (int r = 0; r < 8; ++r) {
      const int row = qb * 16 + r + rbase;
      out[((size_t)batch * S_ + row) * KD_ + col] = oacc[t][r] / lv[r];
    }
  }
}

// ---------------------------------------------------------------------------
// Host launcher.  d_in: x[4,2048,1024] f32, WQ/WK/WV [1024,1024] f32.
// d_out: [4,2048,1024] f32.
// Workspace (f16): Qh 16MB | Kh 16MB | Vt 16MB | xh 16MB | WQh/WKh/WVh 6MB
// ---------------------------------------------------------------------------
extern "C" void kernel_launch(void* const* d_in, const int* in_sizes, int n_in,
                              void* d_out, int out_size, void* d_ws,
                              size_t ws_size, hipStream_t stream) {
  const float* x  = (const float*)d_in[0];
  const float* WQ = (const float*)d_in[1];
  const float* WK = (const float*)d_in[2];
  const float* WV = (const float*)d_in[3];
  float* out = (float*)d_out;

  _Float16* Qh  = (_Float16*)d_ws;
  _Float16* Kh  = Qh  + (size_t)NROW * KD_;
  _Float16* Vt  = Kh  + (size_t)NROW * KD_;
  _Float16* xh  = Vt  + (size_t)NROW * KD_;
  _Float16* WQh = xh  + (size_t)NROW * E_;
  _Float16* WKh = WQh + (size_t)KD_ * E_;
  _Float16* WVh = WKh + (size_t)KD_ * E_;

  // fp32 -> f16 pre-pass (float4-vectorized)
  const int nx4 = (NROW * E_) / 4;       // 2097152
  const int nw4 = (KD_ * E_) / 4;        // 262144
  cvt_f16_kernel<<<dim3((nx4 + 255) / 256), dim3(256), 0, stream>>>(x, xh, nx4);
  cvt_f16_kernel<<<dim3((nw4 + 255) / 256), dim3(256), 0, stream>>>(WQ, WQh, nw4);
  cvt_f16_kernel<<<dim3((nw4 + 255) / 256), dim3(256), 0, stream>>>(WK, WKh, nw4);
  cvt_f16_kernel<<<dim3((nw4 + 255) / 256), dim3(256), 0, stream>>>(WV, WVh, nw4);

  // 3 * 512 * 8 = 12288 wave tasks / 8 waves per block
  qkv_proj_kernel<<<dim3(1536), dim3(256), 0, stream>>>(xh, WQh, WKh, WVh,
                                                        Qh, Kh, Vt);
  // 4 batches * 128 query blocks
  flash_attn_kernel<<<dim3(512), dim3(256), 0, stream>>>(Qh, Kh, Vt, out);
}
